// Self_attention_84464826843815
// MI455X (gfx1250) — matrix operand order repaired and checked by reading, not compiled, not run on hardware
//
#include <hip/hip_runtime.h>

// ---------------------------------------------------------------------------
// Self-attention for B=8, S=2048, D=512 on gfx1250 (MI455X).
// Phase 1: Q/K/V projections via bf16 WMMA GEMM (V stored transposed).
// Phase 2: flash-attention with online softmax, all GEMMs via
//          v_wmma_f32_16x16x32_bf16 (fp32 accumulation).
// ---------------------------------------------------------------------------

typedef __attribute__((ext_vector_type(16))) __bf16 v16bf;
typedef __attribute__((ext_vector_type(8)))  float  v8f;

static constexpr int BATCH = 8;
static constexpr int SEQ   = 2048;
static constexpr int DM    = 512;
static constexpr int MTOT  = BATCH * SEQ;   // 16384 rows

union FragU { v16bf v; uint4 u[2]; };
union Pack4 { __bf16 h[4]; uint2 u; };
union Pack8 { __bf16 h[8]; uint4 u; };

__device__ __forceinline__ v8f vzero() {
  v8f v = {0.f, 0.f, 0.f, 0.f, 0.f, 0.f, 0.f, 0.f};
  return v;
}

// Load a 16-half fragment from two 16-byte-aligned 8-half groups.
__device__ __forceinline__ v16bf load2x16(const __bf16* p0, const __bf16* p1) {
  FragU f;
  f.u[0] = *(const uint4*)p0;
  f.u[1] = *(const uint4*)p1;
  return f.v;
}

// Cross-lane reductions within aligned 8-lane groups via ds_swizzle_b32
// (group-of-32 mode, xor masks 1/2/4; and_mask=0x1f).
__device__ __forceinline__ float swz_max8(float x) {
  x = fmaxf(x, __int_as_float(__builtin_amdgcn_ds_swizzle(__float_as_int(x), 0x041f)));
  x = fmaxf(x, __int_as_float(__builtin_amdgcn_ds_swizzle(__float_as_int(x), 0x081f)));
  x = fmaxf(x, __int_as_float(__builtin_amdgcn_ds_swizzle(__float_as_int(x), 0x101f)));
  return x;
}
__device__ __forceinline__ float swz_sum8(float x) {
  x += __int_as_float(__builtin_amdgcn_ds_swizzle(__float_as_int(x), 0x041f));
  x += __int_as_float(__builtin_amdgcn_ds_swizzle(__float_as_int(x), 0x081f));
  x += __int_as_float(__builtin_amdgcn_ds_swizzle(__float_as_int(x), 0x101f));
  return x;
}

// ---------------------------------------------------------------------------
// Kernel 1: C[M=16384, N=512] = data[M,512] @ W[512,512]  (fp32 in, bf16 out)
// blockIdx.z: 0 -> Q (row major), 1 -> K (row major), 2 -> V (transposed out)
// Tile: 128(M) x 128(N), K-step 32. 256 threads = 8 waves (4 M x 2 N groups);
// each wave computes 32x64 via 2x4 WMMA fragments (8 WMMAs per K-step).
// ---------------------------------------------------------------------------
__global__ __launch_bounds__(256) void proj_kernel(
    const float* __restrict__ data,
    const float* __restrict__ Wq,
    const float* __restrict__ Wk,
    const float* __restrict__ Wv,
    __bf16* __restrict__ Q,
    __bf16* __restrict__ K,
    __bf16* __restrict__ Vt) {
  __shared__ __align__(16) __bf16 As[128][32];  // A tile, row major (8 KB)
  __shared__ __align__(16) __bf16 Bt[128][32];  // W tile transposed: Bt[n][k] (8 KB)

  const int z  = blockIdx.z;
  const float* W = (z == 0) ? Wq : ((z == 1) ? Wk : Wv);
  const int m0 = blockIdx.y * 128;
  const int n0 = blockIdx.x * 128;
  const int tid = threadIdx.x;
  const int lane = tid & 31, wid = tid >> 5;
  const int wm = wid & 3, wn = wid >> 2;      // 4 M-groups x 2 N-groups of waves
  const int hi = lane >> 4, ln = lane & 15;

  v8f acc[2][4];
#pragma unroll
  for (int i = 0; i < 2; ++i)
#pragma unroll
    for (int j = 0; j < 4; ++j) acc[i][j] = vzero();

  for (int k0 = 0; k0 < DM; k0 += 32) {
    // Stage A tile (128x32 fp32 -> bf16). 256 threads * 16 floats.
    {
      const int r  = tid >> 3;          // 0..31
      const int c4 = (tid & 7) * 4;     // 0..28
#pragma unroll
      for (int p = 0; p < 4; ++p) {
        const int row = r + p * 32;
        float4 dv = *(const float4*)&data[(size_t)(m0 + row) * DM + k0 + c4];
        Pack4 pk;
        pk.h[0] = (__bf16)dv.x; pk.h[1] = (__bf16)dv.y;
        pk.h[2] = (__bf16)dv.z; pk.h[3] = (__bf16)dv.w;
        *(uint2*)&As[row][c4] = pk.u;
      }
    }
    // Stage W tile transposed (32x128 fp32 -> Bt[n][k] bf16). 4 float4/thread.
    {
#pragma unroll
      for (int p = 0; p < 4; ++p) {
        const int idx = tid + p * 256;       // 0..1023
        const int r   = idx >> 5;            // k row: 0..31
        const int cg  = (idx & 31) * 4;      // col group: 0..124
        float4 wv = *(const float4*)&W[(size_t)(k0 + r) * DM + n0 + cg];
        Bt[cg + 0][r] = (__bf16)wv.x;
        Bt[cg + 1][r] = (__bf16)wv.y;
        Bt[cg + 2][r] = (__bf16)wv.z;
        Bt[cg + 3][r] = (__bf16)wv.w;
      }
    }
    __syncthreads();

    v16bf a[2], b[4];
#pragma unroll
    for (int i = 0; i < 2; ++i) {
      const int row = wm * 32 + i * 16 + ln;          // A: M = lane&15
      a[i] = load2x16(&As[row][hi * 8], &As[row][hi * 8 + 16]);
    }
#pragma unroll
    for (int j = 0; j < 4; ++j) {
      const int col = wn * 64 + j * 16 + ln;          // B: N = lane&15
      b[j] = load2x16(&Bt[col][hi * 8], &Bt[col][hi * 8 + 16]);
    }
#pragma unroll
    for (int i = 0; i < 2; ++i)
#pragma unroll
      for (int j = 0; j < 4; ++j)
        acc[i][j] = __builtin_amdgcn_wmma_f32_16x16x32_bf16(
            false, a[i], false, b[j], (short)0, acc[i][j], false, false);
    __syncthreads();
  }

  // Epilogue (z-branch hoisted outside the unrolled store loops).
  // C-fragment layout: row = r + 8*hi, col = lane&15.
  if (z != 2) {
    __bf16* __restrict__ O = (z == 0) ? Q : K;
#pragma unroll
    for (int i = 0; i < 2; ++i) {
#pragma unroll
      for (int j = 0; j < 4; ++j) {
        const int col = n0 + wn * 64 + j * 16 + ln;
#pragma unroll
        for (int r = 0; r < 8; ++r) {
          const int row = m0 + wm * 32 + i * 16 + r + hi * 8;
          O[(size_t)row * DM + col] = (__bf16)acc[i][j][r];
        }
      }
    }
  } else {
#pragma unroll
    for (int i = 0; i < 2; ++i) {
#pragma unroll
      for (int j = 0; j < 4; ++j) {
        const int col = n0 + wn * 64 + j * 16 + ln;
#pragma unroll
        for (int r = 0; r < 8; ++r) {
          const int row = m0 + wm * 32 + i * 16 + r + hi * 8;
          const int bidx = row >> 11;          // row / SEQ
          const int s    = row & (SEQ - 1);    // row % SEQ
          Vt[((size_t)bidx * DM + col) * SEQ + s] = (__bf16)acc[i][j][r];
        }
      }
    }
  }
}

// ---------------------------------------------------------------------------
// Kernel 2: causal flash attention, key tile Bc = 64.
// One block = 128 threads (4 waves) handles 16 query rows of one batch.
// Wave w owns D-columns [128w, 128w+128): 4 Q fragments + 8 fp32 accumulators.
// Per 64-key tile: partial QK^T per wave (16 WMMA) -> LDS reduce -> fully
// parallel online softmax (16 rows x 8 lanes, ds_swizzle row reductions) ->
// rescale accumulators -> P@V (16 WMMA) from transposed V.
// ---------------------------------------------------------------------------
__global__ __launch_bounds__(128) void attn_kernel(
    const __bf16* __restrict__ Q,
    const __bf16* __restrict__ K,
    const __bf16* __restrict__ Vt,
    float* __restrict__ out) {
  const int b   = blockIdx.y;
  const int q0  = blockIdx.x * 16;
  const int tid = threadIdx.x;
  const int lane = tid & 31, w = tid >> 5;
  const int hi = lane >> 4, ln = lane & 15;

  __shared__ __align__(16) float  Sred[4][16][64];  // per-wave partial scores (16 KB)
  __shared__ __align__(16) __bf16 P[16][64];        // softmaxed probs, bf16 (2 KB)
  __shared__ __align__(16) float  m_s[16];
  __shared__ __align__(16) float  l_s[16];
  __shared__ __align__(16) float  scl_s[16];

  if (tid < 16) { m_s[tid] = -3.0e38f; l_s[tid] = 0.0f; }

  // Preload this wave's Q strip (16 rows x 128 cols) as 4 A-fragments.
  v16bf qf[4];
  {
    const __bf16* qbase = Q + ((size_t)(b * SEQ + q0 + ln)) * DM;
#pragma unroll
    for (int kk = 0; kk < 4; ++kk) {
      const int d0 = w * 128 + kk * 32 + hi * 8;
      qf[kk] = load2x16(qbase + d0, qbase + d0 + 16);
    }
  }
  v8f acc[8];
#pragma unroll
  for (int f = 0; f < 8; ++f) acc[f] = vzero();

  __syncthreads();

  const float scale = 0.022097086912079608f;  // 1/sqrt(2048) (== sqrt(seq_len))
  const int ntiles = (q0 + 16 + 63) >> 6;     // causal: only tiles with t0 <= q0+15

  // Softmax thread mapping: 16 rows x 8 lanes (each row fully inside one wave).
  const int srow = tid >> 3;       // 0..15
  const int ssub = tid & 7;        // 0..7
  const int sj0  = ssub * 8;       // first of 8 columns owned by this lane

  for (int it = 0; it < ntiles; ++it) {
    const int t0 = it * 64;

    // Speculative prefetches of next K / V^T tiles (RT speculative -> safe).
    if (it + 1 < ntiles) {
      __builtin_prefetch(
          K + ((size_t)(b * SEQ + t0 + 64 + lane)) * DM + w * 128, 0, 0);
      __builtin_prefetch(
          Vt + ((size_t)b * DM + w * 128 + lane) * SEQ + t0 + 64, 0, 0);
    }

    // ---- partial scores: S_w(16x64) = Q_w(16x128) @ K_w^T(128x64) ----
    v8f sz[4];
#pragma unroll
    for (int nt = 0; nt < 4; ++nt) sz[nt] = vzero();
#pragma unroll
    for (int kk = 0; kk < 4; ++kk) {
#pragma unroll
      for (int nt = 0; nt < 4; ++nt) {
        const __bf16* kb = K +
            ((size_t)(b * SEQ + t0 + nt * 16 + ln)) * DM + w * 128 + kk * 32 + hi * 8;
        v16bf bf = load2x16(kb, kb + 16);
        sz[nt] = __builtin_amdgcn_wmma_f32_16x16x32_bf16(
            false, qf[kk], false, bf, (short)0, sz[nt], false, false);
      }
    }
#pragma unroll
    for (int nt = 0; nt < 4; ++nt)
#pragma unroll
      for (int r = 0; r < 8; ++r)
        Sred[w][r + hi * 8][nt * 16 + ln] = sz[nt][r];
    __syncthreads();

    // ---- parallel online softmax: 16 rows x 8 lanes, 8 columns per lane ----
    {
      float s[8];
#pragma unroll
      for (int q = 0; q < 8; ++q) s[q] = 0.0f;
#pragma unroll
      for (int wv = 0; wv < 4; ++wv) {
        float4 a0 = *(const float4*)&Sred[wv][srow][sj0];
        float4 a1 = *(const float4*)&Sred[wv][srow][sj0 + 4];
        s[0] += a0.x; s[1] += a0.y; s[2] += a0.z; s[3] += a0.w;
        s[4] += a1.x; s[5] += a1.y; s[6] += a1.z; s[7] += a1.w;
      }
      const float mold = m_s[srow];     // safe: all lanes of a row share a wave
      float lmax = mold;
#pragma unroll
      for (int q = 0; q < 8; ++q) {
        float v = s[q] * scale;
        if (t0 + sj0 + q > q0 + srow) v = -1.0e6f;  // causal mask (ref: -1e6)
        s[q] = v;
        lmax = fmaxf(lmax, v);
      }
      lmax = swz_max8(lmax);            // row max (includes m_old)
      float lsum = 0.0f;
      Pack8 pk;
#pragma unroll
      for (int q = 0; q < 8; ++q) {
        const float p = __expf(s[q] - lmax);
        lsum += p;
        pk.h[q] = (__bf16)p;
      }
      *(uint4*)&P[srow][sj0] = pk.u;    // one ds_store_b128 per lane
      lsum = swz_sum8(lsum);            // row sum of exp
      if (ssub == 0) {
        const float sclf = __expf(mold - lmax);
        m_s[srow]  = lmax;
        l_s[srow]  = l_s[srow] * sclf + lsum;
        scl_s[srow] = sclf;
      }
    }
    __syncthreads();

    // ---- rescale accumulators by exp(m_old - m_new) per row ----
    {
      float4 s0 = *(const float4*)&scl_s[hi * 8];
      float4 s1 = *(const float4*)&scl_s[hi * 8 + 4];
      const float sc8[8] = {s0.x, s0.y, s0.z, s0.w, s1.x, s1.y, s1.z, s1.w};
#pragma unroll
      for (int f = 0; f < 8; ++f)
#pragma unroll
        for (int r = 0; r < 8; ++r) acc[f][r] *= sc8[r];
    }

    // ---- acc += P(16x64) @ V(64x128_w), B-operand from transposed V ----
    {
      v16bf pf[2];
#pragma unroll
      for (int kk = 0; kk < 2; ++kk)
        pf[kk] = load2x16(&P[ln][kk * 32 + hi * 8], &P[ln][kk * 32 + hi * 8 + 16]);
#pragma unroll
      for (int nt = 0; nt < 8; ++nt) {
        const int d = w * 128 + nt * 16 + ln;
        const __bf16* vb = Vt + ((size_t)b * DM + d) * SEQ + t0 + hi * 8;
#pragma unroll
        for (int kk = 0; kk < 2; ++kk) {
          v16bf vf = load2x16(vb + kk * 32, vb + kk * 32 + 16);
          acc[nt] = __builtin_amdgcn_wmma_f32_16x16x32_bf16(
              false, pf[kk], false, vf, (short)0, acc[nt], false, false);
        }
      }
    }
    __syncthreads();
  }

  // ---- epilogue: out = acc / l  (fp32 output, row = r + 8*hi, col fixed) ----
  {
    float4 l0 = *(const float4*)&l_s[hi * 8];
    float4 l1 = *(const float4*)&l_s[hi * 8 + 4];
    const float li[8] = {l0.x, l0.y, l0.z, l0.w, l1.x, l1.y, l1.z, l1.w};
#pragma unroll
    for (int nt = 0; nt < 8; ++nt) {
      const int d = w * 128 + nt * 16 + ln;
#pragma unroll
      for (int r = 0; r < 8; ++r) {
        const int row = q0 + r + hi * 8;
        out[((size_t)(b * SEQ + row)) * DM + d] = acc[nt][r] / li[r];
      }
    }
  }
}

// ---------------------------------------------------------------------------
// Launch: inputs are fp32 (data, Wq, Wk, Wv); output fp32 [B,S,D].
// Workspace: Q (16 MB bf16) | K (16 MB) | V^T (16 MB).
// ---------------------------------------------------------------------------
extern "C" void kernel_launch(void* const* d_in, const int* in_sizes, int n_in,
                              void* d_out, int out_size, void* d_ws, size_t ws_size,
                              hipStream_t stream) {
  (void)in_sizes; (void)n_in; (void)out_size; (void)ws_size;
  const float* data = (const float*)d_in[0];
  const float* Wq   = (const float*)d_in[1];
  const float* Wk   = (const float*)d_in[2];
  const float* Wv   = (const float*)d_in[3];

  __bf16* Q  = (__bf16*)d_ws;
  __bf16* K  = Q + (size_t)MTOT * DM;
  __bf16* Vt = K + (size_t)MTOT * DM;
  float* out = (float*)d_out;

  dim3 gProj(DM / 128, MTOT / 128, 3);  // (4, 128, 3)
  proj_kernel<<<gProj, 256, 0, stream>>>(data, Wq, Wk, Wv, Q, K, Vt);

  dim3 gAttn(SEQ / 16, BATCH, 1);       // (128, 8)
  attn_kernel<<<gAttn, 128, 0, stream>>>(Q, K, Vt, out);
}
